// RGAT_DGL_28776280883882
// MI455X (gfx1250) — compile-verified, matrix operand order checked by reading, not run on hardware
//
#include <hip/hip_runtime.h>
#include <hip/hip_bf16.h>

// ---------------------------------------------------------------------------
// RGAT (DGL-style) on gfx1250:
//   - f16 WMMA GEMMs with LDS double-buffering and CDNA5 async global->LDS
//     copies tracked by ASYNCcnt (when the builtins are available).
//   - atomic edge softmax for the irregular part.
// ---------------------------------------------------------------------------

typedef __attribute__((ext_vector_type(16))) _Float16 v16h;
typedef __attribute__((ext_vector_type(8)))  _Float16 v8h;
typedef __attribute__((ext_vector_type(4)))  _Float16 v4h;
typedef __attribute__((ext_vector_type(8)))  float    v8f;
typedef __attribute__((ext_vector_type(4)))  float    v4f;
typedef int gv4i __attribute__((vector_size(16)));   // matches builtin pointee

#define INF_   1024
#define HD_    256
#define H_     4
#define D_     64
#define COUT_  1024

#if __has_builtin(__builtin_amdgcn_global_load_async_to_lds_b128) && \
    __has_builtin(__builtin_amdgcn_s_wait_asynccnt)
#define USE_ASYNC_LDS 1
#else
#define USE_ASYNC_LDS 0
#endif

__device__ __forceinline__ void wait_async_all() {
#if USE_ASYNC_LDS
    __builtin_amdgcn_s_wait_asynccnt(0);
#endif
}

// Copy one 16-byte chunk global(f16) -> LDS(f16).
__device__ __forceinline__ void copy16(const _Float16* __restrict__ src,
                                       _Float16* __restrict__ dst)
{
#if USE_ASYNC_LDS
    __builtin_amdgcn_global_load_async_to_lds_b128(
        (__attribute__((address_space(1))) gv4i*)src,
        (__attribute__((address_space(3))) gv4i*)dst, 0, 0);
#else
    *(v8h*)dst = *(const v8h*)src;
#endif
}

// ---------------------------------------------------------------------------
// Stage one K-step tile pair: A 128x32 halves, B^T 64x32 halves.
// A chunks: 512 (2/thread). B chunks: 256 (1/thread). All b128.
// ---------------------------------------------------------------------------
__device__ __forceinline__ void stage_tiles(const _Float16* __restrict__ Ah,
                                            const _Float16* __restrict__ Bt,
                                            _Float16 (&As)[128][40],
                                            _Float16 (&Bs)[64][40],
                                            int tid, int m0, int n0, int k0,
                                            int M, int K)
{
    #pragma unroll
    for (int i = 0; i < 2; ++i) {
        int e  = tid + i * 256;           // 0..511
        int r  = e >> 2, c8 = e & 3;      // row 0..127, 8-half chunk 0..3
        int gm = m0 + r;
        gm = (gm < M) ? gm : (M - 1);     // clamp: rows >= M never stored
        copy16(Ah + (long)gm * K + k0 + c8 * 8, &As[r][c8 * 8]);
    }
    {
        int r  = tid >> 2, c8 = tid & 3;  // col 0..63, chunk 0..3
        copy16(Bt + (long)(n0 + r) * K + k0 + c8 * 8, &Bs[r][c8 * 8]);
    }
}

// ---------------------------------------------------------------------------
// GEMM: C[M,N] = A[M,K] * B[K,N] (+bias); A is f16 row-major [M,K],
// Bt is f16 transposed [N,K]; C is f32. 256 threads = 8 waves; macro tile
// 128x64; each wave computes 16x64 = 4 WMMA tiles sharing one A fragment.
// Requires K % 32 == 0, N % 64 == 0.
// ---------------------------------------------------------------------------
__global__ void __launch_bounds__(256)
wmma_gemm_f16(const _Float16* __restrict__ Ah, const _Float16* __restrict__ Bt,
              const float* __restrict__ bias, float* __restrict__ C,
              int M, int K, int Nn)
{
    __shared__ __align__(16) _Float16 As[2][128][40];
    __shared__ __align__(16) _Float16 Bs[2][64][40];

    const int tid  = threadIdx.x;
    const int lane = tid & 31;
    const int wid  = tid >> 5;
    const int l15  = lane & 15;
    const int kc   = (lane >> 4) * 8;     // 0 (lanes 0-15) / 8 (lanes 16-31)
    const int m0   = blockIdx.x * 128;
    const int n0   = blockIdx.y * 64;

    v8f acc[4];
    #pragma unroll
    for (int j = 0; j < 4; ++j)
        acc[j] = (v8f){0.f,0.f,0.f,0.f,0.f,0.f,0.f,0.f};

    stage_tiles(Ah, Bt, As[0], Bs[0], tid, m0, n0, 0, M, K);
    wait_async_all();
    __syncthreads();

    int cur = 0;
    for (int k0 = 0; k0 < K; k0 += 32) {
        const int nxt = cur ^ 1;
        if (k0 + 32 < K)   // prefetch next tile while computing this one
            stage_tiles(Ah, Bt, As[nxt], Bs[nxt], tid, m0, n0, k0 + 32, M, K);

        // A fragment: row = wid*16 + l15; halves K = kc..kc+7, kc+16..kc+23
        const int arow = wid * 16 + l15;
        v8h alo = *(const v8h*)&As[cur][arow][kc];
        v8h ahi = *(const v8h*)&As[cur][arow][kc + 16];
        v16h afrag = __builtin_shufflevector(alo, ahi,
                        0,1,2,3,4,5,6,7,8,9,10,11,12,13,14,15);

#define WMMA_STEP(J, RA)                                                     \
        {                                                                    \
            const int bcol = (J) * 16 + l15;                                 \
            v8h blo = *(const v8h*)&Bs[cur][bcol][kc];                       \
            v8h bhi = *(const v8h*)&Bs[cur][bcol][kc + 16];                  \
            v16h bfrag = __builtin_shufflevector(blo, bhi,                   \
                            0,1,2,3,4,5,6,7,8,9,10,11,12,13,14,15);          \
            acc[J] = __builtin_amdgcn_wmma_f32_16x16x32_f16(                 \
                        false, afrag, false, bfrag, (short)0, acc[J],        \
                        RA, false);                                          \
        }
        WMMA_STEP(0, true)
        WMMA_STEP(1, true)
        WMMA_STEP(2, true)
        WMMA_STEP(3, false)
#undef WMMA_STEP

        wait_async_all();
        __syncthreads();
        cur = nxt;
    }

    // store: VGPR v -> row = m0 + wid*16 + v + 8*(lane>=16), col = l15
    const int rbase = m0 + wid * 16 + 8 * (lane >> 4);
    #pragma unroll
    for (int j = 0; j < 4; ++j) {
        const int col = n0 + j * 16 + l15;
        const float badd = bias ? bias[col] : 0.f;
        #pragma unroll
        for (int v = 0; v < 8; ++v) {
            const int row = rbase + v;
            if (row < M) C[(long)row * Nn + col] = acc[j][v] + badd;
        }
    }
}

// ---------------------------------------------------------------------------
// f32 -> f16 elementwise convert (vectorized x4)
// ---------------------------------------------------------------------------
__global__ void cvt_f16_kernel(const float* __restrict__ s,
                               _Float16* __restrict__ d, long n)
{
    long i = ((long)blockIdx.x * blockDim.x + threadIdx.x) * 4;
    if (i >= n) return;
    v4f v = *(const v4f*)(s + i);
    v4h h;
    h.x = (_Float16)v.x; h.y = (_Float16)v.y;
    h.z = (_Float16)v.z; h.w = (_Float16)v.w;
    *(v4h*)(d + i) = h;
}

// W[K,N] f32 -> Wt[N,K] f16, LDS-tiled transpose. Grid (N/32, K/32), block 32x8.
__global__ void transpose_w_f16(const float* __restrict__ W,
                                _Float16* __restrict__ Wt, int K, int Nn)
{
    __shared__ float tile[32][33];
    const int n0 = blockIdx.x * 32, k0 = blockIdx.y * 32;
    const int tx = threadIdx.x, ty = threadIdx.y;
    #pragma unroll
    for (int i = 0; i < 32; i += 8)
        tile[ty + i][tx] = W[(long)(k0 + ty + i) * Nn + n0 + tx];
    __syncthreads();
    #pragma unroll
    for (int i = 0; i < 32; i += 8)
        Wt[(long)(n0 + ty + i) * K + k0 + tx] = (_Float16)tile[tx][ty + i];
}

// ---------------------------------------------------------------------------
// Edge-softmax kernels (bandwidth-trivial vs GEMMs)
// ---------------------------------------------------------------------------
__global__ void el_er_kernel(const float* __restrict__ f,
                             const float* __restrict__ al,
                             const float* __restrict__ ar,
                             float* __restrict__ el, float* __restrict__ er,
                             int n /* = N*H */)
{
    int i = blockIdx.x * blockDim.x + threadIdx.x;
    if (i >= n) return;
    int h = i & 3;
    const float* fp = f + (long)(i >> 2) * HD_ + h * D_;
    const float* ap = al + h * D_;
    const float* bp = ar + h * D_;
    float sl = 0.f, sr = 0.f;
    #pragma unroll 8
    for (int d = 0; d < D_; ++d) {
        float v = fp[d];
        sl += v * ap[d];
        sr += v * bp[d];
    }
    el[i] = sl;
    er[i] = sr;
}

__global__ void init_nh_kernel(unsigned* __restrict__ mo, float* __restrict__ ss, int n)
{
    int i = blockIdx.x * blockDim.x + threadIdx.x;
    if (i < n) { mo[i] = 0u; ss[i] = 0.f; }
}

__device__ __forceinline__ unsigned ordu(float v) {
    unsigned u = __float_as_uint(v);
    return (u & 0x80000000u) ? ~u : (u | 0x80000000u);
}
__device__ __forceinline__ float unordu(unsigned u) {
    unsigned b = (u & 0x80000000u) ? (u ^ 0x80000000u) : ~u;
    return __uint_as_float(b);
}

__global__ void edge_logit_max(const int* __restrict__ src, const int* __restrict__ dst,
                               const float* __restrict__ el, const float* __restrict__ er,
                               float* __restrict__ elog, unsigned* __restrict__ mo,
                               int n /* = E*H */)
{
    int i = blockIdx.x * blockDim.x + threadIdx.x;
    if (i >= n) return;
    int e = i >> 2, h = i & 3;
    int s = src[e], d = dst[e];
    float v = el[s * H_ + h] + er[d * H_ + h];
    v = (v > 0.f) ? v : 0.2f * v;           // leaky_relu(0.2)
    elog[i] = v;
    atomicMax(&mo[d * H_ + h], ordu(v));
}

__global__ void edge_exp_sum(const int* __restrict__ dst,
                             float* __restrict__ elog,
                             const unsigned* __restrict__ mo,
                             float* __restrict__ ss, int n /* = E*H */)
{
    int i = blockIdx.x * blockDim.x + threadIdx.x;
    if (i >= n) return;
    int e = i >> 2, h = i & 3;
    int d = dst[e];
    float m = unordu(mo[d * H_ + h]);
    float ex = __expf(elog[i] - m);
    elog[i] = ex;
    atomicAdd(&ss[d * H_ + h], ex);
}

__global__ void edge_aggregate(const int* __restrict__ src, const int* __restrict__ dst,
                               const float* __restrict__ elog, const float* __restrict__ ss,
                               const float* __restrict__ f, float* __restrict__ hnext,
                               int n /* = E*64 */)
{
    int i = blockIdx.x * blockDim.x + threadIdx.x;
    if (i >= n) return;
    int e = i >> 6, d0 = i & 63;
    int s = src[e], dd = dst[e];
    #pragma unroll
    for (int h = 0; h < H_; ++h) {
        float sv = ss[dd * H_ + h];
        float a = elog[e * H_ + h] / (sv > 0.f ? sv : 1.f);
        atomicAdd(&hnext[(long)dd * HD_ + h * D_ + d0],
                  a * f[(long)s * HD_ + h * D_ + d0]);
    }
}

__global__ void init_bias3(const float* __restrict__ b0, const float* __restrict__ b1,
                           const float* __restrict__ b2, float* __restrict__ h, int n)
{
    int i = blockIdx.x * blockDim.x + threadIdx.x;
    if (i >= n) return;
    int j = i & (HD_ - 1);
    h[i] = b0[j] + b1[j] + b2[j];
}

__global__ void lrelu_kernel(float* __restrict__ h, int n)
{
    int i = blockIdx.x * blockDim.x + threadIdx.x;
    if (i >= n) return;
    float v = h[i];
    h[i] = (v > 0.f) ? v : 0.01f * v;
}

// ---------------------------------------------------------------------------
extern "C" void kernel_launch(void* const* d_in, const int* in_sizes, int n_in,
                              void* d_out, int out_size, void* d_ws, size_t ws_size,
                              hipStream_t stream)
{
    const int N = in_sizes[0] / INF_;          // 40000
    const int E = in_sizes[1];                 // 250000

    const float* x = (const float*)d_in[0];
    const int* src[3] = { (const int*)d_in[1], (const int*)d_in[3], (const int*)d_in[5] };
    const int* dst[3] = { (const int*)d_in[2], (const int*)d_in[4], (const int*)d_in[6] };
    auto P = [&](int l, int t, int w) -> const float* {
        return (const float*)d_in[7 + l * 12 + t * 4 + w];
    };
    const float* Wout = (const float*)d_in[31];
    const float* bout = (const float*)d_in[32];

    // ---- workspace layout (all sections 16B-aligned) ----
    char* w = (char*)d_ws;
    const long NND = (long)N * HD_;            // 10.24M floats
    float*    f_buf = (float*)w;              w += NND * 4;
    float*    hA    = (float*)w;              w += NND * 4;
    float*    hB    = (float*)w;              w += NND * 4;
    float*    el    = (float*)w;              w += (long)N * H_ * 4;
    float*    er    = (float*)w;              w += (long)N * H_ * 4;
    float*    ssum  = (float*)w;              w += (long)N * H_ * 4;
    unsigned* mo    = (unsigned*)w;           w += (long)N * H_ * 4;
    float*    elog  = (float*)w;              w += (long)E * H_ * 4;
    _Float16* xh    = (_Float16*)w;           w += (long)N * INF_ * 2;
    _Float16* hh    = (_Float16*)w;           w += NND * 2;
    _Float16* Wt    = (_Float16*)w;           w += (long)INF_ * HD_ * 2;

    const int TB   = 256;
    const int nNH  = N * H_;
    const int nEH  = E * H_;
    const int nED  = E * D_;
    const int nNHD = (int)NND;

    dim3 gemm_grid256((N + 127) / 128, HD_ / 64);
    dim3 gemm_gridOut((N + 127) / 128, COUT_ / 64);
    dim3 tb32x8(32, 8);

    // ---------------- Layer 0 (input x, K=1024) ----------------
    cvt_f16_kernel<<<(int)(((long)N * INF_ / 4 + TB - 1) / TB), TB, 0, stream>>>(
        x, xh, (long)N * INF_);
    init_bias3<<<(nNHD + TB - 1) / TB, TB, 0, stream>>>(P(0,0,3), P(0,1,3), P(0,2,3), hA, nNHD);
    for (int t = 0; t < 3; ++t) {
        transpose_w_f16<<<dim3(HD_ / 32, INF_ / 32), tb32x8, 0, stream>>>(P(0,t,0), Wt, INF_, HD_);
        wmma_gemm_f16<<<gemm_grid256, TB, 0, stream>>>(xh, Wt, nullptr, f_buf, N, INF_, HD_);
        el_er_kernel<<<(nNH + TB - 1) / TB, TB, 0, stream>>>(f_buf, P(0,t,1), P(0,t,2), el, er, nNH);
        init_nh_kernel<<<(nNH + TB - 1) / TB, TB, 0, stream>>>(mo, ssum, nNH);
        edge_logit_max<<<(nEH + TB - 1) / TB, TB, 0, stream>>>(src[t], dst[t], el, er, elog, mo, nEH);
        edge_exp_sum<<<(nEH + TB - 1) / TB, TB, 0, stream>>>(dst[t], elog, mo, ssum, nEH);
        edge_aggregate<<<(nED + TB - 1) / TB, TB, 0, stream>>>(src[t], dst[t], elog, ssum, f_buf, hA, nED);
    }
    lrelu_kernel<<<(nNHD + TB - 1) / TB, TB, 0, stream>>>(hA, nNHD);
    cvt_f16_kernel<<<(int)((NND / 4 + TB - 1) / TB), TB, 0, stream>>>(hA, hh, NND);

    // ---------------- Layer 1 (input hA as f16, K=256) ----------------
    init_bias3<<<(nNHD + TB - 1) / TB, TB, 0, stream>>>(P(1,0,3), P(1,1,3), P(1,2,3), hB, nNHD);
    for (int t = 0; t < 3; ++t) {
        transpose_w_f16<<<dim3(HD_ / 32, HD_ / 32), tb32x8, 0, stream>>>(P(1,t,0), Wt, HD_, HD_);
        wmma_gemm_f16<<<gemm_grid256, TB, 0, stream>>>(hh, Wt, nullptr, f_buf, N, HD_, HD_);
        el_er_kernel<<<(nNH + TB - 1) / TB, TB, 0, stream>>>(f_buf, P(1,t,1), P(1,t,2), el, er, nNH);
        init_nh_kernel<<<(nNH + TB - 1) / TB, TB, 0, stream>>>(mo, ssum, nNH);
        edge_logit_max<<<(nEH + TB - 1) / TB, TB, 0, stream>>>(src[t], dst[t], el, er, elog, mo, nEH);
        edge_exp_sum<<<(nEH + TB - 1) / TB, TB, 0, stream>>>(dst[t], elog, mo, ssum, nEH);
        edge_aggregate<<<(nED + TB - 1) / TB, TB, 0, stream>>>(src[t], dst[t], elog, ssum, f_buf, hB, nED);
    }

    // ---------------- Output projection: logits = hB @ Wout + bout ----------
    cvt_f16_kernel<<<(int)((NND / 4 + TB - 1) / TB), TB, 0, stream>>>(hB, hh, NND);
    transpose_w_f16<<<dim3(COUT_ / 32, HD_ / 32), tb32x8, 0, stream>>>(Wout, Wt, HD_, COUT_);
    wmma_gemm_f16<<<gemm_gridOut, TB, 0, stream>>>(hh, Wt, bout, (float*)d_out, N, HD_, COUT_);
}
